// CSDehaze_17265768530063
// MI455X (gfx1250) — matrix operand order repaired and checked by reading, not compiled, not run on hardware
//
#include <hip/hip_runtime.h>
#include <hip/hip_bf16.h>
#include <math.h>

// ---------------------------------------------------------------------------
// CSDehaze block for MI455X (gfx1250, wave32, WMMA).
// All big GEMMs run on v_wmma_f32_16x16x32_bf16 (bf16 in, f32 accum).
// NCHW->token transpose stages through LDS with global_load_async_to_lds_b32.
// ---------------------------------------------------------------------------

typedef __bf16 bf16_t;
typedef __attribute__((ext_vector_type(16))) __bf16 v16bf;
typedef __attribute__((ext_vector_type(8)))  __bf16 v8bf;
typedef __attribute__((ext_vector_type(8)))  float  v8f;

#define BATCH   4
#define CCH     96
#define HH      256
#define WW      256
#define PLANE   65536              // H*W
#define T_TOK   262144             // B*H*W
#define NELEM   25165824           // B*C*H*W
#define PER_B   6291456            // C*H*W
#define SCALE_F 0.17677669529663687f   // 32^-0.5
#define LOGIT_MAX_F 4.605170185988091f // log(100)

// ---------------- WMMA fragment helpers (per CDNA5 ISA layouts) -------------

__device__ __forceinline__ v16bf mk16(v8bf lo, v8bf hi) {
  union { v16bf v; v8bf h[2]; } u;
  u.h[0] = lo; u.h[1] = hi;
  return u.v;
}

// A (16xK tile, row-major source). Lane m<16: K = kb+0..7 and kb+16..23.
// Lane m>=16: K = kb+8..15 and kb+24..31.  rowptr = &A[row(lane)][0].
__device__ __forceinline__ v16bf ldfragA(const bf16_t* rowptr, int kbase, int lane) {
  int k0 = kbase + ((lane & 16) ? 8 : 0);
  v8bf lo = *(const v8bf*)(rowptr + k0);
  v8bf hi = *(const v8bf*)(rowptr + k0 + 16);
  return mk16(lo, hi);
}

// B (Kx16 tile, column(n)-major source = W[n][k] row-major over k).
// Lane n<16: K = kb+0..7 and kb+16..23; lane n>=16: K = kb+8..15 and kb+24..31.
__device__ __forceinline__ v16bf ldfragB(const bf16_t* colptr, int kbase, int lane) {
  int k0 = kbase + ((lane & 16) ? 8 : 0);
  v8bf lo = *(const v8bf*)(colptr + k0);
  v8bf hi = *(const v8bf*)(colptr + k0 + 16);
  return mk16(lo, hi);
}

#define WMMA_BF16(a, b, c) \
  __builtin_amdgcn_wmma_f32_16x16x32_bf16(false, (a), false, (b), (short)0, (c), false, false)

// window index helpers: wi in [0,4096), r in [0,64)
__device__ __forceinline__ int win_token(int wi, int r) {
  int b  = wi >> 10;
  int rm = wi & 1023;
  int wh = rm >> 5;
  int ww = rm & 31;
  return ((b * 256 + wh * 8 + (r >> 3)) * 256) + ww * 8 + (r & 7);
}

// ---------------------------- small utility kernels -------------------------

__global__ void zero8_k(float* p) {
  if (threadIdx.x < 8) p[threadIdx.x] = 0.0f;
}

__global__ void cast_f2bf_k(const float* __restrict__ s, bf16_t* __restrict__ d, int n) {
  int i = blockIdx.x * 256 + threadIdx.x;
  if (i < n) d[i] = (bf16_t)s[i];
}

__global__ void copyf_k(const float* __restrict__ s, float* __restrict__ d, int n) {
  int i = blockIdx.x * 256 + threadIdx.x;
  if (i < n) d[i] = s[i];
}

// ---------------------------- statistics (AGN) -------------------------------

__global__ void reduce_stats_k(const float* __restrict__ x, float* __restrict__ sums) {
  __shared__ float ss[256];
  __shared__ float sq[256];
  int b = blockIdx.y;
  const float* xb = x + (size_t)b * PER_B + (size_t)blockIdx.x * 6144;
  float s = 0.f, q = 0.f;
  for (int i = threadIdx.x; i < 6144; i += 256) {
    float v = xb[i];
    s += v; q += v * v;
  }
  ss[threadIdx.x] = s; sq[threadIdx.x] = q;
  __syncthreads();
  for (int off = 128; off > 0; off >>= 1) {
    if (threadIdx.x < off) {
      ss[threadIdx.x] += ss[threadIdx.x + off];
      sq[threadIdx.x] += sq[threadIdx.x + off];
    }
    __syncthreads();
  }
  if (threadIdx.x == 0) {
    atomicAdd(&sums[b * 2 + 0], ss[0]);
    atomicAdd(&sums[b * 2 + 1], sq[0]);
  }
}

__global__ void finalize_stats_k(const float* __restrict__ sums,
                                 const float* __restrict__ m1w, const float* __restrict__ m1b,
                                 const float* __restrict__ m2w, const float* __restrict__ m2b,
                                 float* __restrict__ stat2,
                                 float* __restrict__ rescale, float* __restrict__ rebias) {
  int i = threadIdx.x;              // 384 = B*C
  if (i >= BATCH * CCH) return;
  int b = i / CCH, c = i % CCH;
  float cnt  = (float)PER_B;
  float mean = sums[b * 2 + 0] / cnt;
  float var  = sums[b * 2 + 1] / cnt - mean * mean;
  float stdv = sqrtf(var + 1e-5f);
  if (c == 0) { stat2[b * 2] = mean; stat2[b * 2 + 1] = 1.0f / stdv; }
  rescale[i] = stdv * m1w[c] + m1b[c];
  rebias[i]  = mean * m2w[c] + m2b[c];
}

__global__ void xn_k(const float* __restrict__ x, const float* __restrict__ stat2,
                     float* __restrict__ xn) {
  int idx = blockIdx.x * 256 + threadIdx.x;
  if (idx >= NELEM) return;
  int b = (idx >> 16) / CCH;
  xn[idx] = (x[idx] - stat2[b * 2]) * stat2[b * 2 + 1];
}

// -------------------------- depthwise 3x3 branches ---------------------------

__global__ void dw3x3_relu2_k(const float* __restrict__ xn,
                              const float* __restrict__ w1, const float* __restrict__ b1,
                              const float* __restrict__ w2, const float* __restrict__ b2,
                              float* __restrict__ t1, float* __restrict__ t2) {
  int idx = blockIdx.x * 256 + threadIdx.x;
  if (idx >= NELEM) return;
  int w  = idx & 255;
  int h  = (idx >> 8) & 255;
  int pl = idx >> 16;                 // b*96+c
  int c  = pl % CCH;
  const float* base = xn + ((size_t)pl << 16);
  float s1 = 0.f, s2 = 0.f;
  #pragma unroll
  for (int di = 0; di < 3; ++di) {
    int hh = h + di - 1;
    if (hh < 0 || hh > 255) continue;
    #pragma unroll
    for (int dj = 0; dj < 3; ++dj) {
      int ww = w + dj - 1;
      if (ww < 0 || ww > 255) continue;
      float v = base[hh * 256 + ww];
      s1 += v * w1[c * 9 + di * 3 + dj];
      s2 += v * w2[c * 9 + di * 3 + dj];
    }
  }
  s1 += b1[c]; s2 += b2[c];
  t1[idx] = s1 > 0.f ? s1 : 0.f;
  t2[idx] = s2 > 0.f ? s2 : 0.f;
}

__global__ void agn_combine_k(const float* __restrict__ xn,
                              const float* __restrict__ t1, const float* __restrict__ t2,
                              const float* __restrict__ w1, const float* __restrict__ b1,
                              const float* __restrict__ w2, const float* __restrict__ b2,
                              const float* __restrict__ agnw, const float* __restrict__ agnb,
                              const float* __restrict__ rescale, const float* __restrict__ rebias,
                              float* __restrict__ y) {
  int idx = blockIdx.x * 256 + threadIdx.x;
  if (idx >= NELEM) return;
  int w  = idx & 255;
  int h  = (idx >> 8) & 255;
  int pl = idx >> 16;
  int c  = pl % CCH;
  int b  = pl / CCH;
  const float* base1 = t1 + ((size_t)pl << 16);
  const float* base2 = t2 + ((size_t)pl << 16);
  float s1 = 0.f, s2 = 0.f;
  #pragma unroll
  for (int di = 0; di < 3; ++di) {
    int hh = h + di - 1;
    if (hh < 0 || hh > 255) continue;
    #pragma unroll
    for (int dj = 0; dj < 3; ++dj) {
      int ww = w + dj - 1;
      if (ww < 0 || ww > 255) continue;
      s1 += base1[hh * 256 + ww] * w1[c * 9 + di * 3 + dj];
      s2 += base2[hh * 256 + ww] * w2[c * 9 + di * 3 + dj];
    }
  }
  float local   = s1 + b1[c];
  float texture = s2 + b2[c];
  y[idx] = xn[idx] * (agnw[c] * rescale[b * CCH + c])
         + (agnb[c] + rebias[b * CCH + c]) + local + texture;
}

// NCHW fp32 -> token-major [T][96] bf16 (LDS-tiled transpose).
// Global->LDS copy uses CDNA5 async-to-LDS (ASYNCcnt) instead of VGPR staging.
__global__ void nchw_to_tok_bf_k(const float* __restrict__ src, bf16_t* __restrict__ dst) {
  __shared__ float tile[32][33];
  int bz  = blockIdx.z;
  int ct  = blockIdx.y;           // channel tile (3 x 32)
  int hw0 = blockIdx.x * 32;
  #pragma unroll
  for (int i = 0; i < 4; ++i) {
    int cl = threadIdx.y + i * 8;
    const float* g = &src[((size_t)(bz * CCH + ct * 32 + cl) << 16) + hw0 + threadIdx.x];
    unsigned lds = (unsigned)(size_t)&tile[cl][threadIdx.x];
    asm volatile("global_load_async_to_lds_b32 %0, %1, off"
                 :: "v"(lds), "v"(g) : "memory");
  }
  asm volatile("s_wait_asynccnt 0" ::: "memory");
  __syncthreads();
  #pragma unroll
  for (int i = 0; i < 4; ++i) {
    int hl = threadIdx.y + i * 8;
    int t  = (bz << 16) + hw0 + hl;
    dst[(size_t)t * CCH + ct * 32 + threadIdx.x] = (bf16_t)tile[threadIdx.x][hl];
  }
}

// ------------------------ generic WMMA token GEMM ----------------------------
// D[t][n] = sum_k A[t][k] * Wt[n][k] + bias[n], fused epilogues.
// MODE 0: QKV   -> n<96: outf=Q fp32[T][96]; n>=96: outbf=kv bf16[T][192]
// MODE 1: proj  -> x1 = identity_nchw + d*rescale + rebias; outf=x1 f32, outbf=x1 bf16
// MODE 2: mlp1  -> outbf = relu(d) bf16 [T][384]
// MODE 3: mlp2  -> out_nchw = x1f[t][n] + d   (final output, NCHW)
// KK <= 128: A fragments are preloaded once per wave and reused for all N-tiles.
template <int KK, int MODE>
__global__ void gemm_tok_k(const bf16_t* __restrict__ A, int lda,
                           const bf16_t* __restrict__ Wt,
                           const float* __restrict__ bias,
                           int NC,
                           float* __restrict__ outf,
                           bf16_t* __restrict__ outbf,
                           const float* __restrict__ identity,
                           const float* __restrict__ rescale,
                           const float* __restrict__ rebias,
                           const float* __restrict__ x1f,
                           float* __restrict__ outnchw) {
  constexpr int NKC   = KK / 32;
  constexpr bool HOLD = (NKC <= 4);
  int lane    = threadIdx.x & 31;
  int wave    = threadIdx.x >> 5;
  int rowbase = (blockIdx.x * 4 + wave) * 16;
  const bf16_t* arow = A + (size_t)(rowbase + (lane & 15)) * lda;
  int rhalf = (lane & 16) ? 8 : 0;
  int nt_count = NC >> 4;

  v16bf afrag[HOLD ? NKC : 1];
  if constexpr (HOLD) {
    #pragma unroll
    for (int kc = 0; kc < NKC; ++kc) afrag[kc] = ldfragA(arow, kc * 32, lane);
  }

  for (int nt = 0; nt < nt_count; ++nt) {
    const bf16_t* bcol = Wt + (size_t)(nt * 16 + (lane & 15)) * KK;
    v8f acc = {0.f, 0.f, 0.f, 0.f, 0.f, 0.f, 0.f, 0.f};
    #pragma unroll
    for (int kc = 0; kc < NKC; ++kc) {
      v16bf a;
      if constexpr (HOLD) {
        a = afrag[kc];
      } else {
        __builtin_prefetch((const void*)(arow + kc * 32 + 64), 0, 1);
        a = ldfragA(arow, kc * 32, lane);
      }
      v16bf b = ldfragB(bcol, kc * 32, lane);
      acc = WMMA_BF16(a, b, acc);
    }
    int n   = nt * 16 + (lane & 15);
    float bn = bias[n];
    #pragma unroll
    for (int v = 0; v < 8; ++v) {
      int t   = rowbase + v + rhalf;
      float d = acc[v] + bn;
      if constexpr (MODE == 0) {
        if (n < 96) outf[(size_t)t * 96 + n] = d;
        else        outbf[(size_t)t * 192 + (n - 96)] = (bf16_t)d;
      } else if constexpr (MODE == 1) {
        int b_ = t >> 16;
        int hw = t & 65535;
        float xv = identity[((size_t)(b_ * CCH + n) << 16) + hw];
        float x1 = xv + d * rescale[b_ * CCH + n] + rebias[b_ * CCH + n];
        outf [(size_t)t * 96 + n] = x1;
        outbf[(size_t)t * 96 + n] = (bf16_t)x1;
      } else if constexpr (MODE == 2) {
        float r = d > 0.f ? d : 0.f;
        outbf[(size_t)t * 384 + n] = (bf16_t)r;
      } else {
        int b_ = t >> 16;
        int hw = t & 65535;
        outnchw[((size_t)(b_ * CCH + n) << 16) + hw] = x1f[(size_t)t * 96 + n] + d;
      }
    }
  }
}

// ------------------- 5x5 reflect depthwise conv on Q (token-major) ----------

__global__ void dw5x5_reflect_k(const float* __restrict__ Qf,
                                const float* __restrict__ w, const float* __restrict__ b,
                                bf16_t* __restrict__ cwbf) {
  int idx = blockIdx.x * 256 + threadIdx.x;
  if (idx >= NELEM) return;
  int c = idx % 96;
  int t = idx / 96;
  int b_ = t >> 16;
  int hw = t & 65535;
  int h = hw >> 8, x = hw & 255;
  float s = 0.f;
  #pragma unroll
  for (int i = 0; i < 5; ++i) {
    int hr = h + i - 2;
    hr = hr < 0 ? -hr : (hr > 255 ? 510 - hr : hr);
    #pragma unroll
    for (int j = 0; j < 5; ++j) {
      int wr = x + j - 2;
      wr = wr < 0 ? -wr : (wr > 255 ? 510 - wr : wr);
      int tt = (b_ << 16) + hr * 256 + wr;
      s += Qf[(size_t)tt * 96 + c] * w[c * 25 + i * 5 + j];
    }
  }
  cwbf[(size_t)t * 96 + c] = (bf16_t)(s + b[c]);
}

// -------------------- relative position bias MLP table -----------------------

__global__ void bias_mlp_k(const float* __restrict__ w1, const float* __restrict__ b1,
                           const float* __restrict__ w2, const float* __restrict__ b2,
                           float* __restrict__ bias_tbl) {
  int idx = blockIdx.x * 256 + threadIdx.x;   // 4096 = 64*64
  if (idx >= 4096) return;
  int n = idx >> 6, m = idx & 63;
  float d0 = (float)((n >> 3) - (m >> 3));
  float d1 = (float)((n & 7) - (m & 7));
  float r0 = copysignf(log1pf(fabsf(d0)), d0);
  float r1 = copysignf(log1pf(fabsf(d1)), d1);
  float a0 = 0.f, a1 = 0.f, a2 = 0.f;
  for (int k = 0; k < 256; ++k) {
    float hv = r0 * w1[k * 2 + 0] + r1 * w1[k * 2 + 1] + b1[k];
    hv = hv > 0.f ? hv : 0.f;
    a0 += hv * w2[0 * 256 + k];
    a1 += hv * w2[1 * 256 + k];
    a2 += hv * w2[2 * 256 + k];
  }
  bias_tbl[(0 * 64 + n) * 64 + m] = a0 + b2[0];
  bias_tbl[(1 * 64 + n) * 64 + m] = a1 + b2[1];
  bias_tbl[(2 * 64 + n) * 64 + m] = a2 + b2[2];
}

// ----------------------------- window attention ------------------------------
// grid = (4096 windows, 3 heads), block = 128 (4 waves); wave w owns queries
// 16w..16w+15.  S = q k^T (4 WMMAs), row softmax via half-wave shfl reductions,
// P staged in LDS (D-layout -> A-layout reshape), O = P V (4 WMMAs).
__global__ void attn_win_k(const bf16_t* __restrict__ cw,
                           const bf16_t* __restrict__ kv,
                           const float* __restrict__ bias_tbl,
                           const float* __restrict__ logit_scale,
                           bf16_t* __restrict__ obf) {
  __shared__ bf16_t PT[64][72];
  __shared__ bf16_t VT[32][72];
  int wi   = blockIdx.x;
  int head = blockIdx.y;
  int tid  = threadIdx.x;
  int lane = tid & 31;
  int wave = tid >> 5;

  // stage V transposed: VT[d][key]
  for (int idx = tid; idx < 2048; idx += 128) {
    int key = idx >> 5, d = idx & 31;
    int t = win_token(wi, key);
    VT[d][key] = kv[(size_t)t * 192 + 96 + head * 32 + d];
  }
  __syncthreads();

  float ls = __expf(fminf(logit_scale[0], LOGIT_MAX_F));
  float sf = ls * SCALE_F;

  // S = q k^T
  int tq = win_token(wi, wave * 16 + (lane & 15));
  v16bf qf = ldfragA(cw + (size_t)tq * 96 + head * 32, 0, lane);
  v8f S[4];
  #pragma unroll
  for (int nt = 0; nt < 4; ++nt) {
    int tk = win_token(wi, nt * 16 + (lane & 15));
    v16bf kf = ldfragB(kv + (size_t)tk * 192 + head * 32, 0, lane);
    v8f z = {0.f, 0.f, 0.f, 0.f, 0.f, 0.f, 0.f, 0.f};
    S[nt] = WMMA_BF16(qf, kf, z);
  }

  // row softmax (row = VGPR index in D-layout, cols across 16-lane half)
  int rhalf = (lane & 16) ? 8 : 0;
  #pragma unroll
  for (int v = 0; v < 8; ++v) {
    int qi = wave * 16 + v + rhalf;
    const float* brow = bias_tbl + ((size_t)(head * 64 + qi)) * 64 + (lane & 15);
    float e0 = S[0][v] * sf + brow[0];
    float e1 = S[1][v] * sf + brow[16];
    float e2 = S[2][v] * sf + brow[32];
    float e3 = S[3][v] * sf + brow[48];
    float mx = fmaxf(fmaxf(e0, e1), fmaxf(e2, e3));
    for (int msk = 1; msk < 16; msk <<= 1) mx = fmaxf(mx, __shfl_xor(mx, msk, 32));
    e0 = __expf(e0 - mx); e1 = __expf(e1 - mx);
    e2 = __expf(e2 - mx); e3 = __expf(e3 - mx);
    float sm = e0 + e1 + e2 + e3;
    for (int msk = 1; msk < 16; msk <<= 1) sm += __shfl_xor(sm, msk, 32);
    float inv = 1.0f / sm;
    int cb = lane & 15;
    PT[qi][cb +  0] = (bf16_t)(e0 * inv);
    PT[qi][cb + 16] = (bf16_t)(e1 * inv);
    PT[qi][cb + 32] = (bf16_t)(e2 * inv);
    PT[qi][cb + 48] = (bf16_t)(e3 * inv);
  }
  // PT rows of this wave are produced and consumed by the same wave; LDS ops
  // are in-order per wave, no cross-wave dependency -> no barrier needed.

  // O = P V
  #pragma unroll
  for (int dt = 0; dt < 2; ++dt) {
    v8f acc = {0.f, 0.f, 0.f, 0.f, 0.f, 0.f, 0.f, 0.f};
    const bf16_t* par = &PT[wave * 16 + (lane & 15)][0];
    const bf16_t* vbr = &VT[dt * 16 + (lane & 15)][0];
    #pragma unroll
    for (int kc = 0; kc < 64; kc += 32) {
      v16bf a = ldfragA(par, kc, lane);
      v16bf b = ldfragB(vbr, kc, lane);
      acc = WMMA_BF16(a, b, acc);
    }
    #pragma unroll
    for (int v = 0; v < 8; ++v) {
      int qi = wave * 16 + v + rhalf;
      int t  = win_token(wi, qi);
      obf[(size_t)t * 96 + head * 32 + dt * 16 + (lane & 15)] = (bf16_t)acc[v];
    }
  }
}

// ------------------------------- launcher ------------------------------------

extern "C" void kernel_launch(void* const* d_in, const int* in_sizes, int n_in,
                              void* d_out, int out_size, void* d_ws, size_t ws_size,
                              hipStream_t stream) {
  const float* x      = (const float*)d_in[0];
  const float* agnw   = (const float*)d_in[1];
  const float* agnb   = (const float*)d_in[2];
  const float* meta1w = (const float*)d_in[3];
  const float* meta1b = (const float*)d_in[4];
  const float* meta2w = (const float*)d_in[5];
  const float* meta2b = (const float*)d_in[6];
  const float* la1w   = (const float*)d_in[7];
  const float* la1b   = (const float*)d_in[8];
  const float* la2w   = (const float*)d_in[9];
  const float* la2b   = (const float*)d_in[10];
  const float* ta1w   = (const float*)d_in[11];
  const float* ta1b   = (const float*)d_in[12];
  const float* ta2w   = (const float*)d_in[13];
  const float* ta2b   = (const float*)d_in[14];
  const float* q_w    = (const float*)d_in[15];
  const float* q_b    = (const float*)d_in[16];
  const float* kv_w   = (const float*)d_in[17];
  const float* kv_b   = (const float*)d_in[18];
  const float* dw_w   = (const float*)d_in[19];
  const float* dw_b   = (const float*)d_in[20];
  const float* proj_w = (const float*)d_in[21];
  const float* proj_b = (const float*)d_in[22];
  const float* lscale = (const float*)d_in[23];
  const float* rp_w1  = (const float*)d_in[24];
  const float* rp_b1  = (const float*)d_in[25];
  const float* rp_w2  = (const float*)d_in[26];
  const float* rp_b2  = (const float*)d_in[27];
  const float* m1_w   = (const float*)d_in[28];
  const float* m1_b   = (const float*)d_in[29];
  const float* m2_w   = (const float*)d_in[30];
  const float* m2_b   = (const float*)d_in[31];
  float* out = (float*)d_out;

  // ---- workspace layout (regions reused across phases; peak ~606 MB) ----
  char* ws = (char*)d_ws;
  const size_t SZP = (size_t)NELEM * 4;       // 100663296 B, one NCHW fp32 tensor
  float*  sums    = (float*)(ws + 0);
  float*  stat2   = (float*)(ws + 64);
  float*  rescale = (float*)(ws + 256);
  float*  rebias  = (float*)(ws + 2048);
  float*  qkvb    = (float*)(ws + 4096);
  float*  biastbl = (float*)(ws + 8192);
  bf16_t* wqkv    = (bf16_t*)(ws + 65536);    // 288x96
  bf16_t* wproj   = (bf16_t*)(ws + 122880);   // 96x96
  bf16_t* wm1     = (bf16_t*)(ws + 147456);   // 384x96
  bf16_t* wm2     = (bf16_t*)(ws + 229376);   // 96x384
  const size_t R1 = 1u << 20;
  const size_t R2 = R1 + 3 * SZP;
  const size_t R3 = R2 + SZP;
  const size_t R4 = R3 + SZP / 2;
  const size_t R5 = R4 + SZP;
  float*  xn    = (float*)(ws + R1);
  float*  t1    = (float*)(ws + R1 + SZP);
  float*  t2    = (float*)(ws + R1 + 2 * SZP);
  float*  Qf    = (float*)(ws + R1);           // reuses xn (dead)
  float*  x1f   = (float*)(ws + R1 + SZP);     // reuses t1 (dead)
  bf16_t* x1bf  = (bf16_t*)(ws + R1 + 2 * SZP);// reuses t2 (dead)
  float*  ynchw = (float*)(ws + R2);
  bf16_t* obf   = (bf16_t*)(ws + R2);          // reuses ynchw (dead)
  bf16_t* hbf   = (bf16_t*)(ws + R2);          // [T][384] spans R2..R4+SZP/2 (all dead)
  bf16_t* ybf   = (bf16_t*)(ws + R3);
  bf16_t* kvbf  = (bf16_t*)(ws + R4);
  bf16_t* cwbf  = (bf16_t*)(ws + R5);

  const int EW = (NELEM + 255) / 256;          // 98304 elementwise blocks

  // ---- stats + constants ----
  zero8_k<<<1, 32, 0, stream>>>(sums);
  reduce_stats_k<<<dim3(1024, BATCH), 256, 0, stream>>>(x, sums);
  finalize_stats_k<<<1, 384, 0, stream>>>(sums, meta1w, meta1b, meta2w, meta2b,
                                          stat2, rescale, rebias);
  bias_mlp_k<<<16, 256, 0, stream>>>(rp_w1, rp_b1, rp_w2, rp_b2, biastbl);
  cast_f2bf_k<<<(9216 + 255) / 256, 256, 0, stream>>>(q_w, wqkv, 9216);
  cast_f2bf_k<<<(18432 + 255) / 256, 256, 0, stream>>>(kv_w, wqkv + 9216, 18432);
  cast_f2bf_k<<<(9216 + 255) / 256, 256, 0, stream>>>(proj_w, wproj, 9216);
  cast_f2bf_k<<<(36864 + 255) / 256, 256, 0, stream>>>(m1_w, wm1, 36864);
  cast_f2bf_k<<<(36864 + 255) / 256, 256, 0, stream>>>(m2_w, wm2, 36864);
  copyf_k<<<1, 96, 0, stream>>>(q_b, qkvb, 96);
  copyf_k<<<1, 192, 0, stream>>>(kv_b, qkvb + 96, 192);

  // ---- AGN ----
  xn_k<<<EW, 256, 0, stream>>>(x, stat2, xn);
  dw3x3_relu2_k<<<EW, 256, 0, stream>>>(xn, la1w, la1b, ta1w, ta1b, t1, t2);
  agn_combine_k<<<EW, 256, 0, stream>>>(xn, t1, t2, la2w, la2b, ta2w, ta2b,
                                        agnw, agnb, rescale, rebias, ynchw);
  nchw_to_tok_bf_k<<<dim3(2048, 3, BATCH), dim3(32, 8), 0, stream>>>(ynchw, ybf);

  // ---- QKV GEMM (WMMA) ----
  gemm_tok_k<96, 0><<<4096, 128, 0, stream>>>(ybf, 96, wqkv, qkvb, 288,
                                              Qf, kvbf, nullptr, nullptr, nullptr,
                                              nullptr, nullptr);
  // ---- conv-modulated q (5x5 reflect dwconv) ----
  dw5x5_reflect_k<<<EW, 256, 0, stream>>>(Qf, dw_w, dw_b, cwbf);

  // ---- window attention (WMMA) ----
  attn_win_k<<<dim3(4096, 3), 128, 0, stream>>>(cwbf, kvbf, biastbl, lscale, obf);

  // ---- proj GEMM + residual/affine (WMMA) ----
  gemm_tok_k<96, 1><<<4096, 128, 0, stream>>>(obf, 96, wproj, proj_b, 96,
                                              x1f, x1bf, x, rescale, rebias,
                                              nullptr, nullptr);
  // ---- MLP (WMMA) ----
  gemm_tok_k<96, 2><<<4096, 128, 0, stream>>>(x1bf, 96, wm1, m1_b, 384,
                                              nullptr, hbf, nullptr, nullptr, nullptr,
                                              nullptr, nullptr);
  gemm_tok_k<384, 3><<<4096, 128, 0, stream>>>(hbf, 384, wm2, m2_b, 96,
                                               nullptr, nullptr, nullptr, nullptr, nullptr,
                                               x1f, out);
}
